// TransformerDecoderLayerWithCache_84963043049687
// MI455X (gfx1250) — matrix operand order repaired and checked by reading, not compiled, not run on hardware
//
#include <hip/hip_runtime.h>
#include <cstdint>

// ---------------------------------------------------------------------------
// TransformerDecoderLayer for MI455X (gfx1250, wave32, WMMA).
// All GEMMs (QKV, projections, FFN) and both attention contractions run on
// v_wmma_f32_16x16x32_bf16. GEMM tiles are staged with double-buffered
// GLOBAL_LOAD_ASYNC_TO_LDS_B128 (ASYNCcnt-tracked, no VGPR staging).
// ---------------------------------------------------------------------------

#define D_MODEL 1024
#define N_HEADS 16
#define D_HEAD  64
#define D_FF    4096
#define SEQ     2048
#define MEMLEN  1024
#define BATCH   2
#define ROWS    (BATCH * SEQ)     // 4096 query rows total
#define KVROWS  (BATCH * MEMLEN)  // 2048 memory rows total
#define LOG2E   1.4426950408889634f

typedef unsigned short u16;
typedef __attribute__((ext_vector_type(16))) __bf16 v16bf;
typedef __attribute__((ext_vector_type(8)))  float  v8f;

// ----------------------------- helpers -------------------------------------

__device__ inline u16 f2bf(float f) {
  unsigned u = __float_as_uint(f);
  unsigned r = u + 0x7FFFu + ((u >> 16) & 1u);   // round-to-nearest-even
  return (u16)(r >> 16);
}
__device__ inline float bf2f(u16 h) { return __uint_as_float(((unsigned)h) << 16); }

// Build a 16-element bf16 fragment from two 16B chunks (global or LDS).
__device__ inline v16bf load_frag(const u16* p0, const u16* p1) {
  union { struct { uint4 a, b; } u; v16bf f; } c;
  c.u.a = *(const uint4*)p0;
  c.u.b = *(const uint4*)p1;
  return c.f;
}

__device__ inline v8f wmma_bf16(v16bf a, v16bf b, v8f c) {
  // 8 args: (neg_a, A, neg_b, B, c_mod, C, reuse_a, reuse_b)
  return __builtin_amdgcn_wmma_f32_16x16x32_bf16(false, a, false, b, (short)0, c,
                                                 false, false);
}

// Async 16B/lane copy global -> LDS (tracked by ASYNCcnt, no VGPR staging).
// lds = hardware LDS byte address (low 32 bits of the generic LDS pointer).
__device__ inline void async_ld_b128(unsigned lds, const u16* g) {
  asm volatile("global_load_async_to_lds_b128 %0, %1, off"
               :: "v"(lds), "v"(g) : "memory");
}

// Reductions across a 16-lane half of the wave (the WMMA C-layout row group).
__device__ inline float halfmax(float v) {
#pragma unroll
  for (int m = 1; m < 16; m <<= 1) v = fmaxf(v, __shfl_xor(v, m, 32));
  return v;
}
__device__ inline float halfsum(float v) {
#pragma unroll
  for (int m = 1; m < 16; m <<= 1) v += __shfl_xor(v, m, 32);
  return v;
}

// ----------------------------- fp32 -> bf16 convert -------------------------

__global__ __launch_bounds__(256) void f2bf_kernel(const float* __restrict__ in,
                                                   u16* __restrict__ out, int n) {
  int i = blockIdx.x * 256 + threadIdx.x;
  if (i < n) out[i] = f2bf(in[i]);
}

// ----------------------------- LayerNorm ------------------------------------

__global__ __launch_bounds__(256) void ln_bf16_kernel(const float* __restrict__ x,
                                                      const float* __restrict__ g,
                                                      const float* __restrict__ b,
                                                      u16* __restrict__ out) {
  int row = blockIdx.x, tid = threadIdx.x;
  const float* xr = x + (size_t)row * D_MODEL;
  __shared__ float red[256];
  float s = 0.f;
  for (int d = tid; d < D_MODEL; d += 256) s += xr[d];
  red[tid] = s; __syncthreads();
  for (int o = 128; o; o >>= 1) { if (tid < o) red[tid] += red[tid + o]; __syncthreads(); }
  float mu = red[0] * (1.f / D_MODEL);
  __syncthreads();
  float v = 0.f;
  for (int d = tid; d < D_MODEL; d += 256) { float t = xr[d] - mu; v += t * t; }
  red[tid] = v; __syncthreads();
  for (int o = 128; o; o >>= 1) { if (tid < o) red[tid] += red[tid + o]; __syncthreads(); }
  float rs = rsqrtf(red[0] * (1.f / D_MODEL) + 1e-5f);
  u16* orow = out + (size_t)row * D_MODEL;
  for (int d = tid; d < D_MODEL; d += 256)
    orow[d] = f2bf((xr[d] - mu) * rs * g[d] + b[d]);
}

// ----------------------------- block router ---------------------------------
// keep[blk] = sigmoid(relu(mean_k(blk) @ w1^T + b1) @ w2^T + b2) > 0.5
// (batch 0 only, mean over 64 rows of the K part of the fused QKV output)

__global__ __launch_bounds__(128) void router_kernel(const u16* __restrict__ qkv,
                                                     const float* __restrict__ w1,
                                                     const float* __restrict__ b1,
                                                     const float* __restrict__ w2,
                                                     const float* __restrict__ b2,
                                                     int* __restrict__ keep) {
  int blk = blockIdx.x, tid = threadIdx.x;
  __shared__ float meanv[D_MODEL];
  __shared__ float red[128];
  for (int d = tid; d < D_MODEL; d += 128) {
    float s = 0.f;
    for (int r = 0; r < 64; ++r)
      s += bf2f(qkv[(size_t)(blk * 64 + r) * (3 * D_MODEL) + D_MODEL + d]);
    meanv[d] = s * (1.f / 64.f);
  }
  __syncthreads();
  float acc = b1[tid];
  const float* wr = w1 + (size_t)tid * D_MODEL;
  for (int d = 0; d < D_MODEL; ++d) acc += meanv[d] * wr[d];
  red[tid] = fmaxf(acc, 0.f) * w2[tid];
  __syncthreads();
  for (int o = 64; o; o >>= 1) { if (tid < o) red[tid] += red[tid + o]; __syncthreads(); }
  if (tid == 0) keep[blk] = (red[0] + b2[0] > 0.f) ? 1 : 0;
}

// ----------------------------- WMMA GEMM ------------------------------------
// C[M,N] = A[M,K](bf16) @ W[N,K]^T(bf16)  (+bias)(+residual)(+relu)
// 256 threads = 8 waves, 128x128 tile, k-step 32, double-buffered LDS filled
// by global_load_async_to_lds_b128. Wave (mw,nw) owns a 32x64 subtile.

__global__ __launch_bounds__(256) void gemm_bf16_kernel(
    const u16* __restrict__ A, const u16* __restrict__ W,
    int M, int N, int K,
    const float* __restrict__ bias, const float* __restrict__ residual,
    float* __restrict__ outF, u16* __restrict__ outB, int relu) {
  __shared__ __align__(16) u16 As[2][128][40];   // 40-element rows: 80B stride
  __shared__ __align__(16) u16 Bs[2][128][40];
  int tid = threadIdx.x, lane = tid & 31, w = tid >> 5;
  int mw = w >> 1, nw = w & 1;
  int mt = blockIdx.y * 128, nt = blockIdx.x * 128;
  int mloc = lane & 15;
  int kb8  = (lane < 16) ? 0 : 8;    // unused: A fragments also use kb16
  int kb16 = (lane < 16) ? 0 : 16;   // B-fragment K base per half-wave

  // Async staging map: 256 threads x 4 b128 ops cover A(128x32) + B(128x32).
  int r0 = tid >> 2, sg = (tid & 3) * 8;

  v8f acc[2][4] = {};

  auto stage = [&](int buf, int kk) {
    async_ld_b128((unsigned)(uintptr_t)&As[buf][r0][sg],
                  A + (size_t)(mt + r0) * K + kk + sg);
    async_ld_b128((unsigned)(uintptr_t)&As[buf][r0 + 64][sg],
                  A + (size_t)(mt + r0 + 64) * K + kk + sg);
    async_ld_b128((unsigned)(uintptr_t)&Bs[buf][r0][sg],
                  W + (size_t)(nt + r0) * K + kk + sg);
    async_ld_b128((unsigned)(uintptr_t)&Bs[buf][r0 + 64][sg],
                  W + (size_t)(nt + r0 + 64) * K + kk + sg);
  };

  stage(0, 0);
  int nk = K / 32;
  for (int ki = 0; ki < nk; ++ki) {
    int buf = ki & 1;
    if (ki + 1 < nk) {
      stage(buf ^ 1, (ki + 1) * 32);          // prefetch next tile
      asm volatile("s_wait_asynccnt 0x4" ::: "memory");  // current tile landed
    } else {
      asm volatile("s_wait_asynccnt 0x0" ::: "memory");
    }
    __syncthreads();   // all waves' slices resident

    v16bf a0 = load_frag(&As[buf][mw * 32 + mloc][kb16],
                         &As[buf][mw * 32 + mloc][kb16 + 8]);
    v16bf a1 = load_frag(&As[buf][mw * 32 + 16 + mloc][kb16],
                         &As[buf][mw * 32 + 16 + mloc][kb16 + 8]);
#pragma unroll
    for (int j = 0; j < 4; ++j) {
      v16bf bf = load_frag(&Bs[buf][nw * 64 + j * 16 + mloc][kb16],
                           &Bs[buf][nw * 64 + j * 16 + mloc][kb16 + 8]);
      acc[0][j] = wmma_bf16(a0, bf, acc[0][j]);
      acc[1][j] = wmma_bf16(a1, bf, acc[1][j]);
    }
    __syncthreads();   // done reading buf before it is refilled
  }

  // Epilogue: C layout — lane holds column (lane&15); VGPR e holds row e (+8
  // for upper half-wave).
  int rhalf = (lane < 16) ? 0 : 8;
#pragma unroll
  for (int i = 0; i < 2; ++i) {
#pragma unroll
    for (int j = 0; j < 4; ++j) {
      int col = nt + nw * 64 + j * 16 + mloc;
      float bv = bias ? bias[col] : 0.f;
#pragma unroll
      for (int e = 0; e < 8; ++e) {
        int row = mt + mw * 32 + i * 16 + e + rhalf;
        float v = acc[i][j][e] + bv;
        if (residual) v += residual[(size_t)row * N + col];
        if (relu) v = fmaxf(v, 0.f);
        size_t idx = (size_t)row * N + col;
        if (outF) outF[idx] = v;
        if (outB) outB[idx] = f2bf(v);
      }
    }
  }
}

// ----------------------------- flash attention -------------------------------
// One wave per (batch, head, 16-query tile). Online softmax; QK^T and P@V on
// WMMA. Self-attn: causal(-inf) + learned block mask (-1e9). Cross: dense.

__global__ __launch_bounds__(32) void flash_attn_kernel(
    const u16* __restrict__ qb, long qstride,
    const u16* __restrict__ kbp, long kstride,
    const u16* __restrict__ vbp, long vstride,
    u16* __restrict__ ob, int ostride,
    int nq_per_b, int nk_per_b, float scale, int causal,
    const int* __restrict__ keep) {
  int qt = blockIdx.x, h = blockIdx.y, b = blockIdx.z;
  int lane = threadIdx.x;
  int s0q = qt * 16;
  const u16* q  = qb  + (size_t)(b * nq_per_b) * qstride + h * D_HEAD;
  const u16* kp = kbp + (size_t)(b * nk_per_b) * kstride + h * D_HEAD;
  const u16* vp = vbp + (size_t)(b * nk_per_b) * vstride + h * D_HEAD;

  int mloc = lane & 15;
  int rhalf = (lane < 16) ? 0 : 8;
  int kb8  = (lane < 16) ? 0 : 8;
  int kb16 = (lane < 16) ? 0 : 16;

  // Q fragments: A-matrix 16x64 -> two 16x32 bf16 fragments.
  const u16* qr = q + (size_t)(s0q + mloc) * qstride;
  v16bf qf0 = load_frag(qr + kb16,      qr + kb16 + 8);
  v16bf qf1 = load_frag(qr + 32 + kb16, qr + 32 + kb16 + 8);

  v8f o[4] = {};
  float mrow[8], lrow[8];
#pragma unroll
  for (int e = 0; e < 8; ++e) { mrow[e] = -1e30f; lrow[e] = 0.f; }

  __shared__ __align__(16) u16 Pt[16][40];   // probs, A-layout staging
  __shared__ __align__(16) u16 Vt[64][40];   // V^T, B-layout staging

  int nchunk = causal ? (s0q / 32 + 1) : (nk_per_b / 32);
  int qblk = s0q / 64;
  int keepq = keep ? keep[qblk] : 1;

  for (int c = 0; c < nchunk; ++c) {
    int k0 = c * 32;
    bool allowed = keep ? (keepq && keep[k0 / 64]) : true;

    // ---- scores: two 16x16 tiles (keys k0..k0+15, k0+16..k0+31) ------------
    v8f st0 = {}, st1 = {};
    {
      const u16* kr0 = kp + (size_t)(k0 + mloc) * kstride;
      v16bf kA = load_frag(kr0 + kb16,      kr0 + kb16 + 8);        // dh 0..31
      v16bf kB = load_frag(kr0 + 32 + kb16, kr0 + 32 + kb16 + 8);   // dh 32..63
      st0 = wmma_bf16(qf0, kA, st0);
      st0 = wmma_bf16(qf1, kB, st0);
      const u16* kr1 = kp + (size_t)(k0 + 16 + mloc) * kstride;
      v16bf kC = load_frag(kr1 + kb16,      kr1 + kb16 + 8);
      v16bf kD = load_frag(kr1 + 32 + kb16, kr1 + 32 + kb16 + 8);
      st1 = wmma_bf16(qf0, kC, st1);
      st1 = wmma_bf16(qf1, kD, st1);
    }

    // ---- online softmax -----------------------------------------------------
    float alpha[8];
#pragma unroll
    for (int e = 0; e < 8; ++e) {
      int rowg = s0q + e + rhalf;
      int kc0 = k0 + mloc, kc1 = k0 + 16 + mloc;
      float a0 = st0[e] * scale, a1 = st1[e] * scale;
      if (!allowed) { a0 = -1e9f; a1 = -1e9f; }
      if (causal) {
        if (kc0 > rowg) a0 = -3.0e38f;
        if (kc1 > rowg) a1 = -3.0e38f;
      }
      float mx = halfmax(fmaxf(a0, a1));
      float mn = fmaxf(mrow[e], mx);
      float al = exp2f((mrow[e] - mn) * LOG2E);
      float p0 = exp2f((a0 - mn) * LOG2E);
      float p1 = exp2f((a1 - mn) * LOG2E);
      lrow[e] = lrow[e] * al + halfsum(p0 + p1);
      mrow[e] = mn;
      alpha[e] = al;
      st0[e] = p0; st1[e] = p1;
    }
#pragma unroll
    for (int j = 0; j < 4; ++j)
#pragma unroll
      for (int e = 0; e < 8; ++e) o[j][e] *= alpha[e];

    // ---- stage P (A-layout) and V^T (B-layout) through LDS ------------------
#pragma unroll
    for (int e = 0; e < 8; ++e) {
      Pt[rhalf + e][mloc]      = f2bf(st0[e]);
      Pt[rhalf + e][16 + mloc] = f2bf(st1[e]);
    }
    {
      const u16* vr = vp + (size_t)(k0 + lane) * vstride;
#pragma unroll
      for (int d = 0; d < 64; d += 8) {
        union { uint4 q4; u16 u[8]; } cv;
        cv.q4 = *(const uint4*)(vr + d);
#pragma unroll
        for (int t = 0; t < 8; ++t) Vt[d + t][lane] = cv.u[t];
      }
    }
    __syncthreads();   // single-wave block: wave-local LDS fence

    // ---- P @ V: one A-fragment, 4 B-fragments, 4 WMMAs ----------------------
    v16bf pf = load_frag(&Pt[mloc][kb16], &Pt[mloc][kb16 + 8]);
#pragma unroll
    for (int j = 0; j < 4; ++j) {
      v16bf vf = load_frag(&Vt[j * 16 + mloc][kb16], &Vt[j * 16 + mloc][kb16 + 8]);
      o[j] = wmma_bf16(pf, vf, o[j]);
    }
    __syncthreads();
  }

  // ---- normalize and write out (B, S, H*Dh) bf16 ----------------------------
  float rinv[8];
#pragma unroll
  for (int e = 0; e < 8; ++e) rinv[e] = 1.f / fmaxf(lrow[e], 1e-20f);
#pragma unroll
  for (int j = 0; j < 4; ++j) {
    int col = h * D_HEAD + j * 16 + mloc;
#pragma unroll
    for (int e = 0; e < 8; ++e) {
      int row = s0q + e + rhalf;
      ob[(size_t)(b * nq_per_b + row) * ostride + col] = f2bf(o[j][e] * rinv[e]);
    }
  }
}

// ----------------------------- host orchestration ----------------------------

extern "C" void kernel_launch(void* const* d_in, const int* in_sizes, int n_in,
                              void* d_out, int out_size, void* d_ws, size_t ws_size,
                              hipStream_t stream) {
  (void)in_sizes; (void)n_in; (void)out_size; (void)ws_size;
  const float* tgt      = (const float*)d_in[0];
  const float* memory   = (const float*)d_in[1];
  const float* qkv_w    = (const float*)d_in[2];
  const float* sa_out_w = (const float*)d_in[3];
  const float* sa_out_b = (const float*)d_in[4];
  const float* r_w1     = (const float*)d_in[5];
  const float* r_b1     = (const float*)d_in[6];
  const float* r_w2     = (const float*)d_in[7];
  const float* r_b2     = (const float*)d_in[8];
  const float* ca_in_w  = (const float*)d_in[9];
  const float* ca_in_b  = (const float*)d_in[10];
  const float* ca_out_w = (const float*)d_in[11];
  const float* ca_out_b = (const float*)d_in[12];
  const float* ff_w1    = (const float*)d_in[13];
  const float* ff_b1    = (const float*)d_in[14];
  const float* ff_w2    = (const float*)d_in[15];
  const float* ff_b2    = (const float*)d_in[16];
  const float* n1_g = (const float*)d_in[17];
  const float* n1_b = (const float*)d_in[18];
  const float* n2_g = (const float*)d_in[19];
  const float* n2_b = (const float*)d_in[20];
  const float* n3_g = (const float*)d_in[21];
  const float* n3_b = (const float*)d_in[22];
  float* outp = (float*)d_out;

  char* ws = (char*)d_ws;
  size_t off = 0;
  auto alloc = [&](size_t bytes) -> void* {
    void* p = ws + off;
    off += (bytes + 255) & ~(size_t)255;
    return p;
  };
  // ~172 MB total workspace
  u16* wqkv   = (u16*)alloc((size_t)3 * D_MODEL * D_MODEL * 2);
  u16* wsa    = (u16*)alloc((size_t)D_MODEL * D_MODEL * 2);
  u16* wcain  = (u16*)alloc((size_t)3 * D_MODEL * D_MODEL * 2);
  u16* wcaout = (u16*)alloc((size_t)D_MODEL * D_MODEL * 2);
  u16* wff1   = (u16*)alloc((size_t)D_FF * D_MODEL * 2);
  u16* wff2   = (u16*)alloc((size_t)D_MODEL * D_FF * 2);
  u16* membf  = (u16*)alloc((size_t)KVROWS * D_MODEL * 2);
  u16* xbf    = (u16*)alloc((size_t)ROWS * D_MODEL * 2);      // reused ln1/2/3
  u16* qkvbf  = (u16*)alloc((size_t)ROWS * 3 * D_MODEL * 2);
  int* keep   = (int*)alloc(32 * sizeof(int));
  u16* attnbf = (u16*)alloc((size_t)ROWS * D_MODEL * 2);
  float* tgt1 = (float*)alloc((size_t)ROWS * D_MODEL * 4);
  u16* qca    = (u16*)alloc((size_t)ROWS * D_MODEL * 2);
  u16* kvca   = (u16*)alloc((size_t)KVROWS * 2 * D_MODEL * 2);
  u16* cobf   = (u16*)alloc((size_t)ROWS * D_MODEL * 2);
  float* tgt2 = (float*)alloc((size_t)ROWS * D_MODEL * 4);
  u16* ffh    = (u16*)alloc((size_t)ROWS * D_FF * 2);

  auto conv = [&](const float* src, u16* dst, int n) {
    f2bf_kernel<<<(n + 255) / 256, 256, 0, stream>>>(src, dst, n);
  };
  auto gemm = [&](const u16* A, const u16* W, int M, int N, int K,
                  const float* bias, const float* resid,
                  float* outF, u16* outB, int relu) {
    gemm_bf16_kernel<<<dim3(N / 128, M / 128), 256, 0, stream>>>(
        A, W, M, N, K, bias, resid, outF, outB, relu);
  };

  // 0) weight / memory conversions to bf16
  conv(qkv_w,    wqkv,   3 * D_MODEL * D_MODEL);
  conv(sa_out_w, wsa,    D_MODEL * D_MODEL);
  conv(ca_in_w,  wcain,  3 * D_MODEL * D_MODEL);
  conv(ca_out_w, wcaout, D_MODEL * D_MODEL);
  conv(ff_w1,    wff1,   D_FF * D_MODEL);
  conv(ff_w2,    wff2,   D_MODEL * D_FF);
  conv(memory,   membf,  KVROWS * D_MODEL);

  // 1) x = LN1(tgt); qkv = x @ qkv_w^T
  ln_bf16_kernel<<<ROWS, 256, 0, stream>>>(tgt, n1_g, n1_b, xbf);
  gemm(xbf, wqkv, ROWS, 3 * D_MODEL, D_MODEL, nullptr, nullptr, nullptr, qkvbf, 0);

  // 2) block router (batch 0 K means)
  router_kernel<<<32, 128, 0, stream>>>(qkvbf, r_w1, r_b1, r_w2, r_b2, keep);

  // 3) block-sparse causal self-attention
  flash_attn_kernel<<<dim3(SEQ / 16, N_HEADS, BATCH), 32, 0, stream>>>(
      qkvbf,               (long)(3 * D_MODEL),                   // Q
      qkvbf + D_MODEL,     (long)(3 * D_MODEL),                   // K
      qkvbf + 2 * D_MODEL, (long)(3 * D_MODEL),                   // V
      attnbf, D_MODEL, SEQ, SEQ,
      0.03125f /* 1/sqrt(d_model) */, 1, keep);

  // 4) tgt1 = tgt + attn @ sa_out_w^T + b
  gemm(attnbf, wsa, ROWS, D_MODEL, D_MODEL, sa_out_b, tgt, tgt1, nullptr, 0);

  // 5) x2 = LN2(tgt1); cross-attn projections
  ln_bf16_kernel<<<ROWS, 256, 0, stream>>>(tgt1, n2_g, n2_b, xbf);
  gemm(xbf,   wcain,                              ROWS,   D_MODEL,     D_MODEL,
       ca_in_b,           nullptr, nullptr, qca, 0);
  gemm(membf, wcain + (size_t)D_MODEL * D_MODEL,  KVROWS, 2 * D_MODEL, D_MODEL,
       ca_in_b + D_MODEL, nullptr, nullptr, kvca, 0);

  // 6) dense cross-attention (keys = memory, 1/sqrt(head_dim))
  flash_attn_kernel<<<dim3(SEQ / 16, N_HEADS, BATCH), 32, 0, stream>>>(
      qca,            (long)D_MODEL,
      kvca,           (long)(2 * D_MODEL),
      kvca + D_MODEL, (long)(2 * D_MODEL),
      cobf, D_MODEL, SEQ, MEMLEN,
      0.125f /* 1/sqrt(64) */, 0, nullptr);

  // 7) tgt2 = tgt1 + co @ ca_out_w^T + b
  gemm(cobf, wcaout, ROWS, D_MODEL, D_MODEL, ca_out_b, tgt1, tgt2, nullptr, 0);

  // 8) FFN: out = tgt2 + relu(LN3(tgt2) @ w1^T + b1) @ w2^T + b2
  ln_bf16_kernel<<<ROWS, 256, 0, stream>>>(tgt2, n3_g, n3_b, xbf);
  gemm(xbf, wff1, ROWS, D_FF,    D_MODEL, ff_b1, nullptr, nullptr, ffh, 1);
  gemm(ffh, wff2, ROWS, D_MODEL, D_FF,    ff_b2, tgt2,    outp,    nullptr, 0);
}
